// MHA_21165598835581
// MI455X (gfx1250) — compile-verified, matrix operand order checked by reading, not run on hardware
//
#include <hip/hip_runtime.h>
#include <hip/hip_bf16.h>

// ---- problem constants (match reference) ----
#define HIDDEN   2048
#define NHEADS   16
#define HDIM     128
#define SEQ      2048
#define NBATCH   2
#define ROTH     16          // ROTARY_DIM/2
#define NEGINF   (-10000.0f)
#define LOG1E4   9.210340371976184f   // ln(10000)

typedef __attribute__((ext_vector_type(16))) __bf16 v16bf;
typedef __attribute__((ext_vector_type(8)))  __bf16 v8bf;
typedef __attribute__((ext_vector_type(8)))  float  v8f;
typedef __attribute__((ext_vector_type(4)))  float  v4f;
typedef __attribute__((ext_vector_type(4)))  unsigned int v4u;
typedef __attribute__((ext_vector_type(8)))  int    v8i;
typedef __attribute__((ext_vector_type(4)))  int    v4i;

union V16U { v16bf v; v8bf h[2]; };

__device__ __forceinline__ v8f zero8() {
    v8f z;
    #pragma unroll
    for (int i = 0; i < 8; ++i) z[i] = 0.0f;
    return z;
}

#if defined(__has_builtin)
#if __has_builtin(__builtin_amdgcn_tensor_load_to_lds) && __has_builtin(__builtin_amdgcn_s_wait_tensorcnt)
#define USE_TDM 1
#endif
#endif

#ifdef USE_TDM
// ---------------------------------------------------------------------
// Hand-rolled Tensor-DMA descriptor (CDNA5 ISA), 2-D tile,
// bf16 elements (data_size=1 -> 2B). LDS padding generated by the TDM:
//   pad_interval code c -> insert pad every 2^c 8-byte units
//   pad_amount  code a -> pad of (a+1) DWORDs
// This toolchain exposes the 6-arg builtin:
//   (uint32x4 g0, int32x8 g1, int32x4 g2, int32x4 g3, int32x8, i32 cpol)
// ---------------------------------------------------------------------
__device__ __forceinline__ void tdm_load_2d(const void* gptr, unsigned lds_addr,
                                            unsigned tensor_d0, unsigned tensor_d1,
                                            unsigned tile_d0, unsigned tile_d1,
                                            unsigned long long stride0_elems,
                                            unsigned pad_interval, unsigned pad_amount)
{
    unsigned long long ga = (unsigned long long)(uintptr_t)gptr;
    v4u g0;
    g0[0] = 1u;                                        // count=1, user-mode, no gather
    g0[1] = lds_addr;                                  // LDS byte address
    g0[2] = (unsigned)(ga & 0xFFFFFFFFu);              // global_addr[31:0]
    g0[3] = (unsigned)((ga >> 32) & 0x01FFFFFFu)       // global_addr[56:32]
          | 0x80000000u;                               // type = 2 ("image")
    v8i g1;
    g1[0] = (int)((1u << 16)                           // data_size = 2 bytes
                | (1u << 20)                           // pad_enable
                | (pad_interval << 22)
                | (pad_amount   << 25));
    g1[1] = (int)((tensor_d0 & 0xFFFFu) << 16);        // abar_addr=0 | dim0[15:0]
    g1[2] = (int)((tensor_d0 >> 16) | ((tensor_d1 & 0xFFFFu) << 16));
    g1[3] = (int)((tensor_d1 >> 16) | (tile_d0 << 16));
    g1[4] = (int)(tile_d1 & 0xFFFFu);                  // tile_dim2 = 0
    g1[5] = (int)(unsigned)(stride0_elems & 0xFFFFFFFFull);
    g1[6] = (int)(unsigned)((stride0_elems >> 32) & 0xFFFFull); // stride1 = 0
    g1[7] = 0;
    v4i z4 = {0, 0, 0, 0};
    v8i z8 = {0, 0, 0, 0, 0, 0, 0, 0};
    __builtin_amdgcn_tensor_load_to_lds(g0, g1, z4, z4, z8, 0);
}
#endif

// =====================================================================
// Kernel 1: qkv = x @ W^T + b, RoPE on q/k, scatter to bf16 workspace
//   q,k -> [b, h, s, d]   v -> [b, h, d, s]
// Double-buffered LDS; fp32->bf16 conversion happens in the staging path.
// =====================================================================
__global__ __launch_bounds__(256)
void qkv_rope_kernel(const float* __restrict__ x,
                     const float* __restrict__ W,
                     const float* __restrict__ bias,
                     __bf16* __restrict__ qws,
                     __bf16* __restrict__ kws,
                     __bf16* __restrict__ vtws)
{
    __shared__ __align__(16) __bf16 lA[2][128 * 40];   // pitch 80B, conflict-free b128
    __shared__ __align__(16) __bf16 lB[2][128 * 40];

    const int nBase = blockIdx.x * 128;
    const int mBase = blockIdx.y * 128;

    const int tid  = threadIdx.x;
    const int wave = tid >> 5;
    const int lane = tid & 31;
    const int l16  = lane & 15;
    const int kh   = lane >> 4;

    const int rowS = tid >> 1;      // staging row 0..127
    const int half = tid & 1;       // 16-elem half

    v8f acc[8];
    #pragma unroll
    for (int i = 0; i < 8; ++i) acc[i] = zero8();

    // ---- staging lambda: 128x32 fp32 tile -> bf16 LDS ----
    auto stage = [&](int buf, int kk) {
        {
            const float* src = x + (size_t)(mBase + rowS) * HIDDEN + kk + half * 16;
            __bf16* dst = &lA[buf][rowS * 40 + half * 16];
            #pragma unroll
            for (int i = 0; i < 4; ++i) {
                v4f f = *(const v4f*)(src + i * 4);
                dst[i * 4 + 0] = (__bf16)f.x;
                dst[i * 4 + 1] = (__bf16)f.y;
                dst[i * 4 + 2] = (__bf16)f.z;
                dst[i * 4 + 3] = (__bf16)f.w;
            }
        }
        {
            const float* src = W + (size_t)(nBase + rowS) * HIDDEN + kk + half * 16;
            __bf16* dst = &lB[buf][rowS * 40 + half * 16];
            #pragma unroll
            for (int i = 0; i < 4; ++i) {
                v4f f = *(const v4f*)(src + i * 4);
                dst[i * 4 + 0] = (__bf16)f.x;
                dst[i * 4 + 1] = (__bf16)f.y;
                dst[i * 4 + 2] = (__bf16)f.z;
                dst[i * 4 + 3] = (__bf16)f.w;
            }
        }
    };

    stage(0, 0);

    int it = 0;
    for (int kk = 0; kk < HIDDEN; kk += 32, it ^= 1) {
        __syncthreads();                     // buffer `it` ready for everyone
        if (kk + 32 < HIDDEN) stage(it ^ 1, kk + 32);   // overlap next-tile staging

        const __bf16* aptr = &lA[it][(wave * 16 + l16) * 40 + kh * 8];
        V16U a;
        a.h[0] = *(const v8bf*)(aptr);
        a.h[1] = *(const v8bf*)(aptr + 16);

        // grouped fragment loads -> batched back-to-back WMMAs
        #pragma unroll
        for (int g = 0; g < 2; ++g) {
            V16U bb[4];
            #pragma unroll
            for (int j = 0; j < 4; ++j) {
                const __bf16* bptr = &lB[it][((g * 4 + j) * 16 + l16) * 40 + kh * 8];
                bb[j].h[0] = *(const v8bf*)(bptr);
                bb[j].h[1] = *(const v8bf*)(bptr + 16);
            }
            #pragma unroll
            for (int j = 0; j < 4; ++j)
                acc[g * 4 + j] = __builtin_amdgcn_wmma_f32_16x16x32_bf16(
                    false, a.v, false, bb[j].v, (short)0, acc[g * 4 + j], false, false);
        }
    }

    // ---- epilogue: bias + RoPE + scatter ----
    const int qkv_id = nBase >> 11;
    const int head   = (nBase & 2047) >> 7;
    const float invf = __expf(-((float)l16 / (float)ROTH) * LOG1E4);

    #pragma unroll
    for (int r = 0; r < 8; ++r) {
        const int gm = mBase + wave * 16 + r + 8 * kh;
        const int b  = gm >> 11;
        const int s  = gm & (SEQ - 1);

        float vals[8];
        #pragma unroll
        for (int nb = 0; nb < 8; ++nb)
            vals[nb] = acc[nb][r] + bias[nBase + nb * 16 + l16];

        if (qkv_id < 2) {
            float sn, cs;
            __sincosf((float)s * invf, &sn, &cs);
            const float u1 = vals[0], u2 = vals[1];
            vals[0] = u1 * cs - u2 * sn;
            vals[1] = u1 * sn + u2 * cs;
        }

        const size_t bh = (size_t)(b * NHEADS + head);
        if (qkv_id == 0) {
            __bf16* dst = qws + (bh * SEQ + s) * HDIM;
            #pragma unroll
            for (int nb = 0; nb < 8; ++nb) dst[nb * 16 + l16] = (__bf16)vals[nb];
        } else if (qkv_id == 1) {
            __bf16* dst = kws + (bh * SEQ + s) * HDIM;
            #pragma unroll
            for (int nb = 0; nb < 8; ++nb) dst[nb * 16 + l16] = (__bf16)vals[nb];
        } else {
            #pragma unroll
            for (int nb = 0; nb < 8; ++nb)
                vtws[(bh * HDIM + nb * 16 + l16) * SEQ + s] = (__bf16)vals[nb];
        }
    }
}

// =====================================================================
// Kernel 2: causal flash attention. Double-buffered K/V tiles staged by
// the Tensor Data Mover (TENSORcnt-tracked) while waves run WMMAs.
// =====================================================================
__global__ __launch_bounds__(256)
void attn_kernel(const __bf16* __restrict__ qws,
                 const __bf16* __restrict__ kws,
                 const __bf16* __restrict__ vtws,
                 float* __restrict__ out)
{
    __shared__ __align__(16) __bf16 lK[2][32 * 136];   // [key][d], pitch 272B
    __shared__ __align__(16) __bf16 lV[2][128 * 40];   // [d][key], pitch 80B
    __shared__ __align__(16) __bf16 lP[8][16 * 40];    // per-wave P staging

    const int qt = blockIdx.x;
    const int bh = blockIdx.y;
    const int b  = bh >> 4;
    const int h  = bh & 15;

    const int tid  = threadIdx.x;
    const int wave = tid >> 5;
    const int lane = tid & 31;
    const int l16  = lane & 15;
    const int kh   = lane >> 4;

    const int qrowW = qt * 128 + wave * 16;

    // ---- tile staging: TDM (preferred) or synchronous copy fallback ----
    auto stage_tiles = [&](int buf, int kc) {
#ifdef USE_TDM
        if (wave == 0) {
            // K tile: 32 keys x 128 d (row = 256B), pad 16B -> pitch 272B
            tdm_load_2d(kws + ((size_t)bh * SEQ + kc) * HDIM,
                        (unsigned)(uintptr_t)(&lK[buf][0]),
                        /*tensor_d0=*/HDIM, /*tensor_d1=*/32,
                        /*tile_d0=*/HDIM,   /*tile_d1=*/32,
                        /*stride0=*/HDIM,
                        /*pad_interval=*/5, /*pad_amount=*/3);
            // V tile: 128 d x 32 keys (row = 64B), pad 16B -> pitch 80B
            tdm_load_2d(vtws + (size_t)bh * HDIM * SEQ + kc,
                        (unsigned)(uintptr_t)(&lV[buf][0]),
                        /*tensor_d0=*/32,  /*tensor_d1=*/HDIM,
                        /*tile_d0=*/32,    /*tile_d1=*/HDIM,
                        /*stride0=*/SEQ,
                        /*pad_interval=*/3, /*pad_amount=*/3);
        }
#else
        {
            const int key  = tid >> 3;
            const int part = tid & 7;
            const __bf16* src = kws + ((size_t)bh * SEQ + (kc + key)) * HDIM + part * 16;
            __bf16* dst = &lK[buf][key * 136 + part * 16];
            *(v8bf*)dst       = *(const v8bf*)src;
            *(v8bf*)(dst + 8) = *(const v8bf*)(src + 8);
        }
        {
            const int d  = tid >> 1;
            const int hf = tid & 1;
            const __bf16* src = vtws + ((size_t)bh * HDIM + d) * SEQ + kc + hf * 16;
            __bf16* dst = &lV[buf][d * 40 + hf * 16];
            *(v8bf*)dst       = *(const v8bf*)src;
            *(v8bf*)(dst + 8) = *(const v8bf*)(src + 8);
        }
#endif
    };

    // ---- Q fragments: 16 rows x 128 d in registers ----
    V16U qa[4];
    {
        const __bf16* qbase = qws + ((size_t)bh * SEQ + (qrowW + l16)) * HDIM;
        #pragma unroll
        for (int dc = 0; dc < 4; ++dc) {
            qa[dc].h[0] = *(const v8bf*)(qbase + dc * 32 + kh * 8);
            qa[dc].h[1] = *(const v8bf*)(qbase + dc * 32 + 16 + kh * 8);
        }
    }

    v8f o[8];
    #pragma unroll
    for (int i = 0; i < 8; ++i) o[i] = zero8();
    float m_i[8], l_i[8];
    #pragma unroll
    for (int r = 0; r < 8; ++r) { m_i[r] = -1e30f; l_i[r] = 0.0f; }

    const float scale = 0.088388347648318447f;   // 1/sqrt(128)
    const int   kend  = qt * 128 + 128;          // causal bound

    stage_tiles(0, 0);

    int it = 0;
    for (int kc = 0; kc < kend; kc += 32, it ^= 1) {
#ifdef USE_TDM
        if (wave == 0) __builtin_amdgcn_s_wait_tensorcnt(0);
#endif
        __syncthreads();                              // tile `it` visible to all
        if (kc + 32 < kend) stage_tiles(it ^ 1, kc + 32);   // overlap next DMA

        // ---- scores S(16x32) = Q . K^T ----
        v8f sc[2];
        sc[0] = zero8(); sc[1] = zero8();
        #pragma unroll
        for (int nb = 0; nb < 2; ++nb) {
            V16U kb[4];
            #pragma unroll
            for (int dc = 0; dc < 4; ++dc) {
                const __bf16* bptr = &lK[it][(nb * 16 + l16) * 136 + dc * 32 + kh * 8];
                kb[dc].h[0] = *(const v8bf*)(bptr);
                kb[dc].h[1] = *(const v8bf*)(bptr + 16);
            }
            #pragma unroll
            for (int dc = 0; dc < 4; ++dc)
                sc[nb] = __builtin_amdgcn_wmma_f32_16x16x32_bf16(
                             false, qa[dc].v, false, kb[dc].v, (short)0, sc[nb], false, false);
        }

        // ---- online softmax ----
        #pragma unroll
        for (int r = 0; r < 8; ++r) {
            const int qi = qrowW + r + 8 * kh;
            const int k0 = kc + l16;
            const int k1 = kc + 16 + l16;
            float x0 = sc[0][r] * scale + (k0 > qi ? NEGINF : 0.0f);
            float x1 = sc[1][r] * scale + (k1 > qi ? NEGINF : 0.0f);

            float mx = fmaxf(x0, x1);
            #pragma unroll
            for (int m = 1; m < 16; m <<= 1) mx = fmaxf(mx, __shfl_xor(mx, m, 32));
            const float mnew = fmaxf(m_i[r], mx);

            const float p0 = __expf(x0 - mnew);
            const float p1 = __expf(x1 - mnew);
            float rs = p0 + p1;
            #pragma unroll
            for (int m = 1; m < 16; m <<= 1) rs += __shfl_xor(rs, m, 32);

            const float corr = __expf(m_i[r] - mnew);
            l_i[r] = l_i[r] * corr + rs;
            m_i[r] = mnew;
            #pragma unroll
            for (int db = 0; db < 8; ++db) o[db][r] *= corr;

            __bf16* prow = &lP[wave][(r + 8 * kh) * 40];
            prow[l16]      = (__bf16)p0;
            prow[16 + l16] = (__bf16)p1;
        }

        // ---- O += P . V ----
        V16U pa;
        {
            const __bf16* pbase = &lP[wave][l16 * 40];
            pa.h[0] = *(const v8bf*)(pbase + kh * 8);
            pa.h[1] = *(const v8bf*)(pbase + 16 + kh * 8);
        }
        #pragma unroll
        for (int g = 0; g < 2; ++g) {
            V16U vb[4];
            #pragma unroll
            for (int j = 0; j < 4; ++j) {
                const __bf16* vp = &lV[it][((g * 4 + j) * 16 + l16) * 40 + kh * 8];
                vb[j].h[0] = *(const v8bf*)(vp);
                vb[j].h[1] = *(const v8bf*)(vp + 16);
            }
            #pragma unroll
            for (int j = 0; j < 4; ++j)
                o[g * 4 + j] = __builtin_amdgcn_wmma_f32_16x16x32_bf16(
                    false, pa.v, false, vb[j].v, (short)0, o[g * 4 + j], false, false);
        }
    }

    // ---- normalize + write out [b, s, h*128 + d] fp32 ----
    #pragma unroll
    for (int r = 0; r < 8; ++r) {
        const int gq = qrowW + r + 8 * kh;
        const float inv = 1.0f / l_i[r];
        float* dst = out + ((size_t)b * SEQ + gq) * HIDDEN + h * HDIM;
        #pragma unroll
        for (int db = 0; db < 8; ++db)
            dst[db * 16 + l16] = o[db][r] * inv;
    }
}

// =====================================================================
extern "C" void kernel_launch(void* const* d_in, const int* in_sizes, int n_in,
                              void* d_out, int out_size, void* d_ws, size_t ws_size,
                              hipStream_t stream)
{
    const float* x    = (const float*)d_in[0];
    const float* W    = (const float*)d_in[1];
    const float* bias = (const float*)d_in[2];
    float* out = (float*)d_out;

    const size_t per = (size_t)NBATCH * NHEADS * SEQ * HDIM;   // 8Mi elems
    __bf16* qws  = (__bf16*)d_ws;
    __bf16* kws  = qws + per;
    __bf16* vtws = kws + per;

    qkv_rope_kernel<<<dim3(48, 32), 256, 0, stream>>>(x, W, bias, qws, kws, vtws);
    attn_kernel<<<dim3(16, 32), 256, 0, stream>>>(qws, kws, vtws, out);
}
